// CPL_MoE_44839458570560
// MI455X (gfx1250) — matrix operand order repaired and checked by reading, not compiled, not run on hardware
//
#include <hip/hip_runtime.h>
#include <math.h>

// ---- problem constants ----
#define BB 32
#define LL 64
#define HH 1024
#define EE 8
#define VV 12000
#define OUTD 16
#define HID 512   // H/2
#define TV 128    // v-tile per block

typedef __attribute__((ext_vector_type(16))) __bf16 bf16x16;
typedef __attribute__((ext_vector_type(8)))  float  v8f;
typedef __attribute__((ext_vector_type(8)))  unsigned short ushort8;
typedef __attribute__((ext_vector_type(16))) unsigned short ushort16;

__device__ __forceinline__ unsigned short f2bf(float f) {
  unsigned int u = __float_as_uint(f);
  u += 0x7FFFu + ((u >> 16) & 1u);   // round-to-nearest-even
  return (unsigned short)(u >> 16);
}

__device__ __forceinline__ bf16x16 frag16(const unsigned short* p0, const unsigned short* p1) {
  ushort8 lo = *(const ushort8*)p0;
  ushort8 hi = *(const ushort8*)p1;
  ushort16 t;
#pragma unroll
  for (int i = 0; i < 8; ++i) { t[i] = lo[i]; t[i + 8] = hi[i]; }
  return __builtin_bit_cast(bf16x16, t);
}

// ---------------- gate MLP: hid = relu(qr @ w1 + b1) ----------------
__global__ __launch_bounds__(256) void k_hid(const float* __restrict__ qr,
                                             const float* __restrict__ w1,
                                             const float* __restrict__ b1,
                                             float* __restrict__ hid) {
  int idx = blockIdx.x * 256 + threadIdx.x;     // 32*512 = 16384
  int b = idx >> 9, j = idx & 511;
  float acc = b1[j];
  const float* q = qr + b * HH;
#pragma unroll 4
  for (int i = 0; i < HH; ++i) acc += q[i] * w1[i * HID + j];
  hid[idx] = acc > 0.f ? acc : 0.f;
}

// ---- logits, top-2 softmax gates, importance / moe_loss (one block) ----
__global__ __launch_bounds__(256) void k_gate(const float* __restrict__ hid,
                                              const float* __restrict__ w2,
                                              const float* __restrict__ b2,
                                              float* __restrict__ gv,
                                              int* __restrict__ ei,
                                              float* __restrict__ loss_out) {
  __shared__ float logits[BB][EE];
  __shared__ float gates[BB][EE];
  __shared__ float imp[EE];
  int t = threadIdx.x;
  {
    int b = t >> 3, e = t & 7;                  // 256 = 32*8
    float acc = b2[e];
    const float* hb = hid + b * HID;
    for (int i = 0; i < HID; ++i) acc += hb[i] * w2[i * EE + e];
    logits[b][e] = acc;
  }
  __syncthreads();
  if (t < BB) {
    int b = t;
    int i0 = 0; float v0 = logits[b][0];
    for (int e = 1; e < EE; ++e) { float v = logits[b][e]; if (v > v0) { v0 = v; i0 = e; } }
    int i1 = (i0 == 0) ? 1 : 0; float v1 = logits[b][i1];
    for (int e = 0; e < EE; ++e) {
      if (e == i0 || e == i1) continue;
      float v = logits[b][e]; if (v > v1) { v1 = v; i1 = e; }
    }
    float e1 = __expf(v1 - v0);
    float s = 1.0f + e1;
    float g0 = 1.0f / s, g1 = e1 / s;
    ei[b * 2] = i0; ei[b * 2 + 1] = i1;
    gv[b * 2] = g0; gv[b * 2 + 1] = g1;
    for (int e = 0; e < EE; ++e) gates[b][e] = 0.f;
    gates[b][i0] = g0; gates[b][i1] = g1;
  }
  __syncthreads();
  if (t < EE) {
    float s = 0.f;
    for (int b = 0; b < BB; ++b) s += gates[b][t];
    imp[t] = s;
  }
  __syncthreads();
  if (t == 0) {
    float mean = 0.f;
    for (int e = 0; e < EE; ++e) mean += imp[e];
    mean *= (1.0f / EE);
    float var = 0.f;
    for (int e = 0; e < EE; ++e) { float d = imp[e] - mean; var += d * d; }
    var *= (1.0f / (EE - 1));                   // ddof=1
    *loss_out = sqrtf(var) / (mean + 1e-10f) * 0.1f;
  }
}

// ---- gauss head: sigmoid(sum_topk g * (h_last . gauss_w[e,o] + gauss_b[e,o])) ----
__global__ __launch_bounds__(256) void k_gauss(const float* __restrict__ h,
                                               const float* __restrict__ gw,
                                               const float* __restrict__ gb,
                                               const int* __restrict__ ei,
                                               const float* __restrict__ gv,
                                               float* __restrict__ out) {
  int idx = blockIdx.x * 256 + threadIdx.x;
  if (idx >= BB * OUTD) return;
  int b = idx >> 4, o = idx & 15;
  const float* hl = h + ((size_t)b * LL + (LL - 1)) * HH;   // h[:, -1]
  float acc = 0.f;
#pragma unroll
  for (int k = 0; k < 2; ++k) {
    int e = ei[b * 2 + k];
    float g = gv[b * 2 + k];
    const float* w = gw + ((size_t)e * OUTD + o) * HH;
    float d = gb[e * OUTD + o];
    for (int i = 0; i < HH; ++i) d += hl[i] * w[i];
    acc += g * d;
  }
  out[idx] = 1.0f / (1.0f + __expf(-acc));
}

// ---- comp: per (b, v-tile) block; gate-mixed W tile, bf16 WMMA, fp32 accum ----
__global__ __launch_bounds__(256) void k_comp(const float* __restrict__ h,
                                              const float* __restrict__ cw,
                                              const float* __restrict__ cb,
                                              const int* __restrict__ ei,
                                              const float* __restrict__ gv,
                                              float* __restrict__ out) {
  __shared__ __align__(16) unsigned short lds_a[LL * 32];   // 64x32 bf16 (4KB)
  __shared__ __align__(16) unsigned short lds_b[TV * 32];   // 128x32 bf16 (8KB)

  const int b  = blockIdx.y;
  const int v0 = blockIdx.x * TV;
  const int e0 = ei[b * 2], e1 = ei[b * 2 + 1];
  const float g0 = gv[b * 2], g1 = gv[b * 2 + 1];

  const float* __restrict__ hB = h + (size_t)b * LL * HH;
  const float* __restrict__ W0 = cw + (size_t)e0 * VV * HH;
  const float* __restrict__ W1 = cw + (size_t)e1 * VV * HH;

  const int t    = threadIdx.x;
  const int lane = t & 31;
  const int wid  = t >> 5;                  // 8 waves
  // staging assignments
  const int am = t >> 2, ag = (t & 3) * 8;  // A: row 0..63, 8-col group
  const int bn = t >> 1, bg = (t & 1) * 16; // B: row 0..127, 16-col half
  const int vb = v0 + bn;
  const bool bvalid = vb < VV;
  // fragment gather parameters (wave32 layouts)
  const int klo = (lane >> 4) * 16;         // A K-base: 0 / 16
  const int kb  = (lane >> 4) * 16;         // B K-base: 0 / 16
  const int nl  = lane & 15;

  v8f cacc[4];
#pragma unroll
  for (int i = 0; i < 4; ++i) cacc[i] = (v8f){0.f, 0.f, 0.f, 0.f, 0.f, 0.f, 0.f, 0.f};

  for (int kc = 0; kc < HH; kc += 32) {
    // ---- stage A = h[b][64 x 32] as bf16 ----
    {
      const float4* p = (const float4*)(hB + (size_t)am * HH + kc + ag);
      float4 x = p[0], y = p[1];
      ushort8 s;
      s[0] = f2bf(x.x); s[1] = f2bf(x.y); s[2] = f2bf(x.z); s[3] = f2bf(x.w);
      s[4] = f2bf(y.x); s[5] = f2bf(y.y); s[6] = f2bf(y.z); s[7] = f2bf(y.w);
      *(ushort8*)&lds_a[am * 32 + ag] = s;
    }
    // ---- stage B = (g0*W[e0] + g1*W[e1])[128 x 32] as bf16 ----
    {
      const size_t rowOff = (size_t)vb * HH + kc;
      if (bvalid && kc + 32 < HH) {         // prefetch next K-chunk of streamed weights
        __builtin_prefetch((const void*)(W0 + rowOff + 32), 0, 1);
        __builtin_prefetch((const void*)(W1 + rowOff + 32), 0, 1);
      }
#pragma unroll
      for (int half = 0; half < 2; ++half) {
        int col = bg + half * 8;
        float4 a0 = {0, 0, 0, 0}, a1 = {0, 0, 0, 0}, c0 = {0, 0, 0, 0}, c1 = {0, 0, 0, 0};
        if (bvalid) {
          const float4* q0 = (const float4*)(W0 + rowOff + col);
          const float4* q1 = (const float4*)(W1 + rowOff + col);
          a0 = q0[0]; a1 = q0[1]; c0 = q1[0]; c1 = q1[1];
        }
        ushort8 s;
        s[0] = f2bf(g0 * a0.x + g1 * c0.x);
        s[1] = f2bf(g0 * a0.y + g1 * c0.y);
        s[2] = f2bf(g0 * a0.z + g1 * c0.z);
        s[3] = f2bf(g0 * a0.w + g1 * c0.w);
        s[4] = f2bf(g0 * a1.x + g1 * c1.x);
        s[5] = f2bf(g0 * a1.y + g1 * c1.y);
        s[6] = f2bf(g0 * a1.z + g1 * c1.z);
        s[7] = f2bf(g0 * a1.w + g1 * c1.w);
        *(ushort8*)&lds_b[bn * 32 + col] = s;
      }
    }
    __syncthreads();

    // ---- WMMA: each wave = 16 N-cols, 4 M-tiles ----
    const unsigned short* pb = &lds_b[(wid * 16 + nl) * 32 + kb];
    bf16x16 fb = frag16(pb, pb + 8);
#pragma unroll
    for (int mt = 0; mt < 4; ++mt) {
      const unsigned short* pa = &lds_a[(mt * 16 + nl) * 32 + klo];
      bf16x16 fa = frag16(pa, pa + 8);
      cacc[mt] = __builtin_amdgcn_wmma_f32_16x16x32_bf16(
          false, fa, false, fb, (short)0, cacc[mt], false, false);
    }
    __syncthreads();
  }

  // ---- epilogue: add gate-mixed bias, store ----
  const int vcol = v0 + wid * 16 + nl;
  if (vcol < VV) {
    float bias = g0 * cb[e0 * VV + vcol] + g1 * cb[e1 * VV + vcol];
    float* ob = out + (size_t)b * LL * VV + vcol;
    const int rbase = (lane >> 4) * 8;
#pragma unroll
    for (int mt = 0; mt < 4; ++mt) {
#pragma unroll
      for (int r = 0; r < 8; ++r) {
        int row = mt * 16 + rbase + r;
        ob[(size_t)row * VV] = cacc[mt][r] + bias;
      }
    }
  }
}

extern "C" void kernel_launch(void* const* d_in, const int* in_sizes, int n_in,
                              void* d_out, int out_size, void* d_ws, size_t ws_size,
                              hipStream_t stream) {
  const float* qr  = (const float*)d_in[0];
  const float* h   = (const float*)d_in[1];
  const float* gw1 = (const float*)d_in[2];
  const float* gb1 = (const float*)d_in[3];
  const float* gw2 = (const float*)d_in[4];
  const float* gb2 = (const float*)d_in[5];
  const float* gaw = (const float*)d_in[6];
  const float* gab = (const float*)d_in[7];
  const float* cw  = (const float*)d_in[8];
  const float* cb  = (const float*)d_in[9];

  float* out       = (float*)d_out;
  float* gauss_out = out;                                   // 32*16
  float* comp_out  = out + BB * OUTD;                       // 32*64*12000
  float* loss_out  = out + BB * OUTD + (size_t)BB * LL * VV;

  float* wsf = (float*)d_ws;
  float* hid = wsf;                       // 32*512
  float* gvv = wsf + BB * HID;            // 32*2
  int*   ei  = (int*)(wsf + BB * HID + BB * 2);

  k_hid  <<<dim3((BB * HID) / 256), dim3(256), 0, stream>>>(qr, gw1, gb1, hid);
  k_gate <<<dim3(1),                dim3(256), 0, stream>>>(hid, gw2, gb2, gvv, ei, loss_out);
  k_gauss<<<dim3(2),                dim3(256), 0, stream>>>(h, gaw, gab, ei, gvv, gauss_out);
  dim3 grid((VV + TV - 1) / TV, BB);
  k_comp <<<grid,                   dim3(256), 0, stream>>>(h, cw, cb, ei, gvv, comp_out);
}
